// OneODEEncoder_32349693674173
// MI455X (gfx1250) — matrix operand order repaired and checked by reading, not compiled, not run on hardware
//
#include <hip/hip_runtime.h>
#include <hip/hip_bf16.h>
#include <stdint.h>

// Problem sizes (fixed by the reference)
#define H_    1024
#define D_    256
#define T_    256
#define B_    16
#define HD_   1280            // H + D
#define KTOT  20480           // HD * B  (flattened contraction axis)
#define NKB   640             // KTOT / 32 (K-blocks per row-tile)
#define NWG   64              // one workgroup per 16-row output tile
#define NT    512             // threads per workgroup (16 waves, wave32)
#define NWAVE 16
#define KPW   40              // K-blocks per wave (NKB / NWAVE)

typedef __attribute__((ext_vector_type(16))) __bf16 v16bf;
typedef __attribute__((ext_vector_type(8)))  float  v8f;

__device__ __forceinline__ unsigned short f2bf(float f) {
  unsigned u = __float_as_uint(f);
  u += 0x7fffu + ((u >> 16) & 1u);          // round-to-nearest-even
  return (unsigned short)(u >> 16);
}

__device__ __forceinline__ float fast_tanh(float x) {
  float e = __expf(2.0f * x);               // hardware v_exp_f32
  return __fdividef(e - 1.0f, e + 1.0f);
}

// Monotonic-counter grid barrier (64 resident workgroups).
__device__ __forceinline__ void grid_barrier(unsigned* ctr, unsigned target) {
  __builtin_amdgcn_fence(__ATOMIC_RELEASE, "agent");
  __syncthreads();
  if (threadIdx.x == 0) {
    atomicAdd(ctr, 1u);
    while (__hip_atomic_load(ctr, __ATOMIC_RELAXED, __HIP_MEMORY_SCOPE_AGENT) < target)
      __builtin_amdgcn_s_sleep(1);
  }
  __syncthreads();
  __builtin_amdgcn_fence(__ATOMIC_ACQUIRE, "agent");
}

__global__ void ode_init(unsigned* ctr) { *ctr = 0u; }

__global__ __launch_bounds__(NT) void ode_persistent(
    const float* __restrict__ xb,     const float* __restrict__ lift_w,
    const float* __restrict__ lift_b, const float* __restrict__ Wsrc,
    const float* __restrict__ alpha,  const float* __restrict__ beta,
    const float* __restrict__ gain,   const float* __restrict__ bias,
    float* __restrict__ out, unsigned* __restrict__ ctr,
    float* __restrict__ htmp_g, unsigned short* __restrict__ wbf)
{
  __shared__ float hx_s[HD_];                               // [h, x(t)]
  __shared__ __align__(32) unsigned short z_s[KTOT];        // bf16 z vector
  __shared__ float red_s[NWAVE * 16];                       // K-split partials
  __shared__ float alpha_s[B_], beta_s[B_];

  const int tid  = threadIdx.x;
  const int wg   = blockIdx.x;
  const int lane = tid & 31;
  const int wave = tid >> 5;

  if (tid < B_) { alpha_s[tid] = alpha[tid]; beta_s[tid] = beta[tid]; }

  // ---- Phase A: W (f32, [H][KTOT]) -> bf16, pre-swizzled into the
  //      contiguous-k (k = 16*half + j) 16-bit A-matrix (16x32) layout used here: one contiguous
  //      32B chunk per lane per WMMA.
  {
    const unsigned total = (unsigned)H_ * KTOT;
    for (unsigned u = (unsigned)(wg * NT + tid); u < total; u += (unsigned)(NWG * NT)) {
      unsigned j    = u & 15u;          // half index within lane
      unsigned ls   = (u >> 4) & 31u;   // lane slot
      unsigned blk  = u >> 9;           // (row-tile, k-block)
      unsigned kblk = blk % NKB;
      unsigned rt   = blk / NKB;
      unsigned m    = rt * 16u + (ls & 15u);
      unsigned g    = ls >> 4;
      unsigned kk   = j + 16u * g;
      unsigned k    = kblk * 32u + kk;
      wbf[u] = f2bf(Wsrc[(size_t)m * KTOT + k]);
    }
  }

  // ---- Phase A2: h0 = lift_w @ x0 + lift_b. Threads 0..15 own one row
  //      each for the whole integration (RK state lives in registers).
  float h_r = 0.f, k1 = 0.f, k2 = 0.f, k3 = 0.f, gain_r = 0.f, bias_r = 0.f;
  const int row = wg * 16 + tid;     // valid when tid < 16
  if (tid < 16) {
    float s = lift_b[row];
    for (int d = 0; d < D_; ++d) s += lift_w[row * D_ + d] * xb[d];
    h_r = s; gain_r = gain[row]; bias_r = bias[row];
    htmp_g[row] = h_r;
  }

  unsigned nbar = 1;
  grid_barrier(ctr, NWG * nbar); nbar++;

  const float dt = 1.0f / 255.0f;
  const v16bf* __restrict__ Wt = reinterpret_cast<const v16bf*>(wbf);

  for (int step = 0; step < T_ - 1; ++step) {
    const float t0 = step * dt;
    for (int stage = 0; stage < 4; ++stage) {
      const float teval = t0 + ((stage == 0) ? 0.f : (stage == 3) ? dt : 0.5f * dt);
      float uu = teval * 255.0f;
      int i0 = (int)uu; if (i0 > 254) i0 = 254; if (i0 < 0) i0 = 0;
      float w = uu - (float)i0;

      // hx = [htmp, lerp(x, t)]
      for (int i = tid; i < HD_; i += NT) {
        if (i < H_) hx_s[i] = htmp_g[i];
        else {
          int d = i - H_;
          hx_s[i] = (1.0f - w) * xb[i0 * D_ + d] + w * xb[(i0 + 1) * D_ + d];
        }
      }
      __syncthreads();

      // z[k] = tanh(alpha[b] * hx[i] + beta[b])   (k = i*16 + b), bf16
      for (int k = tid; k < KTOT; k += NT) {
        int i = k >> 4, b = k & 15;
        z_s[k] = f2bf(fast_tanh(alpha_s[b] * hx_s[i] + beta_s[b]));
      }
      __syncthreads();

      // WMMA GEMV: phi[16 rows of this wg] = W_tile . z  (z broadcast
      // into all 16 B-columns; A streams W at full bandwidth)
      v8f acc = {};
      const int kb0 = wave * KPW;
      #pragma unroll 4
      for (int kb = 0; kb < KPW; ++kb) {
        int kblk = kb0 + kb;
        v16bf a = Wt[(size_t)(wg * NKB + kblk) * 32 + lane];
        v16bf b = *reinterpret_cast<const v16bf*>(&z_s[kblk * 32 + ((lane >> 4) << 4)]);
        acc = __builtin_amdgcn_wmma_f32_16x16x32_bf16(
            /*neg_a=*/false, a, /*neg_b=*/false, b,
            /*c_mod=*/(short)0, acc, /*reuse_a=*/false, /*reuse_b=*/false);
      }
      // All D columns are identical; lanes 0/16 hold rows 0..7 / 8..15.
      if (lane == 0 || lane == 16) {
        int g = lane >> 4;
        #pragma unroll
        for (int v = 0; v < 8; ++v) red_s[wave * 16 + g * 8 + v] = acc[v];
      }
      __syncthreads();

      if (tid < 16) {
        float s = 0.f;
        #pragma unroll
        for (int wv = 0; wv < NWAVE; ++wv) s += red_s[wv * 16 + tid];
        float fv = fast_tanh(s) * gain_r + bias_r;
        float ht;
        if (stage == 0)      { k1 = fv; ht = h_r + dt * 0.5f * fv; }
        else if (stage == 1) { k2 = fv; ht = h_r + dt * 0.5f * fv; }
        else if (stage == 2) { k3 = fv; ht = h_r + dt * fv; }
        else { h_r += (dt / 6.0f) * (k1 + 2.f * k2 + 2.f * k3 + fv); ht = h_r; }
        htmp_g[row] = ht;
      }
      grid_barrier(ctr, NWG * nbar); nbar++;
    }
  }
  if (tid < 16) out[row] = h_r;
}

extern "C" void kernel_launch(void* const* d_in, const int* in_sizes, int n_in,
                              void* d_out, int out_size, void* d_ws, size_t ws_size,
                              hipStream_t stream) {
  const float* xb     = (const float*)d_in[0];   // (1,T,D)
  const float* lift_w = (const float*)d_in[1];   // (H,D)
  const float* lift_b = (const float*)d_in[2];   // (H,)
  const float* W      = (const float*)d_in[3];   // (H,HD,B)
  const float* alpha  = (const float*)d_in[4];   // (B,)
  const float* beta   = (const float*)d_in[5];   // (B,)
  const float* gain   = (const float*)d_in[6];   // (H,)
  const float* bias   = (const float*)d_in[7];   // (H,)
  float* out = (float*)d_out;                    // (1,H) f32

  char* ws = (char*)d_ws;
  unsigned*       ctr  = (unsigned*)ws;                         // barrier counter
  float*          htmp = (float*)(ws + 256);                    // 1024 f32
  unsigned short* wbf  = (unsigned short*)(ws + 8192);          // ~42 MB bf16 W

  ode_init<<<1, 1, 0, stream>>>(ctr);
  ode_persistent<<<NWG, NT, 0, stream>>>(xb, lift_w, lift_b, W, alpha, beta,
                                         gain, bias, out, ctr, htmp, wbf);
}
